// TFLlamaAttention_56152402428460
// MI455X (gfx1250) — matrix operand order repaired and checked by reading, not compiled, not run on hardware
//
#include <hip/hip_runtime.h>
#include <cstdint>

#define HIDDEN 2048
#define NHEADS 32
#define HDIM 64
#define BATCH 2
#define SEQ 2048
#define MROWS (BATCH * SEQ) // 4096

typedef __attribute__((ext_vector_type(16))) __bf16 v16bf;
typedef __attribute__((ext_vector_type(8)))  float  v8f;
typedef __attribute__((ext_vector_type(4)))  int    v4i;

union FragAB { v16bf v; unsigned short u[16]; uint4 q[2]; };
union FragC  { v8f v; float f[8]; };

__device__ inline unsigned short f2bf(float x) {
    union { float f; unsigned int i; } c; c.f = x;
    unsigned int r = c.i + 0x7FFFu + ((c.i >> 16) & 1u); // round-to-nearest-even
    return (unsigned short)(r >> 16);
}

__device__ inline v8f wmma_bf16(v16bf a, v16bf b, v8f c) {
    // D(16x16,f32) = A(16x32,bf16) * B(32x16,bf16) + C
    return __builtin_amdgcn_wmma_f32_16x16x32_bf16(
        /*neg_a=*/false, a, /*neg_b=*/false, b,
        /*c_mod=*/(short)0, c, /*reuse_a=*/false, /*reuse_b=*/false);
}

// --- async global->LDS copies (CDNA5), with safe fallback ------------------
#if defined(__has_builtin)
#  if __has_builtin(__builtin_amdgcn_global_load_async_to_lds_b128) && \
      __has_builtin(__builtin_amdgcn_s_wait_asynccnt)
#    define USE_ASYNC 1
#  endif
#endif
#ifndef USE_ASYNC
#  define USE_ASYNC 0
#endif

#if USE_ASYNC
// signature (from hipcc diagnostic): (v4i AS(1)*, v4i AS(3)*, imm offset, imm cpol)
typedef __attribute__((address_space(1))) v4i v4i_g;
typedef __attribute__((address_space(3))) v4i v4i_l;
__device__ inline void async_cp16(const unsigned short* g, unsigned short* l) {
    __builtin_amdgcn_global_load_async_to_lds_b128(
        (v4i_g*)(g), (v4i_l*)(l), 0, 0);
}
#  define ASYNC_WAIT() __builtin_amdgcn_s_wait_asynccnt(0)
#else
#  define ASYNC_WAIT()
#endif

// LDS tile row: 32 bf16 payload padded to 72 (144B: 16B-aligned, 36-dword
// stride -> distinct banks across 16 lanes). Double-buffered: rows [buf*128+r].
#define LROW 72

// stage one 128x32 bf16 A tile and one 128x32 bf16 B^T tile (pure copies;
// both operands pre-converted/pre-transposed to bf16 [row][k] layout).
__device__ inline void stage_tiles(const unsigned short* __restrict__ gA,
                                   const unsigned short* __restrict__ gB,
                                   int m0, int n0, int kk, int buf, int tid,
                                   unsigned short (*As)[LROW],
                                   unsigned short (*Bs)[LROW])
{
    #pragma unroll
    for (int hlf = 0; hlf < 2; ++hlf) {
        int c = tid + hlf * 256;             // 512 16B-chunks per tile
        int row = c >> 2, part = (c & 3) * 8;
#if USE_ASYNC
        async_cp16(gA + (size_t)(m0 + row) * HIDDEN + kk + part, &As[buf * 128 + row][part]);
        async_cp16(gB + (size_t)(n0 + row) * HIDDEN + kk + part, &Bs[buf * 128 + row][part]);
#else
        *(uint4*)&As[buf * 128 + row][part] =
            *(const uint4*)(gA + (size_t)(m0 + row) * HIDDEN + kk + part);
        *(uint4*)&Bs[buf * 128 + row][part] =
            *(const uint4*)(gB + (size_t)(n0 + row) * HIDDEN + kk + part);
#endif
    }
}

// ---------------------------------------------------------------------------
// Pre-pass A: X f32 [4096,2048] -> bf16 (same layout)
// ---------------------------------------------------------------------------
__global__ __launch_bounds__(256)
void convert_x_kernel(const float* __restrict__ X, unsigned short* __restrict__ Xb)
{
    size_t i = ((size_t)blockIdx.x * 256 + threadIdx.x) * 8;
    const float4* p = (const float4*)(X + i);
    float4 f0 = p[0], f1 = p[1];
    FragAB t;
    t.u[0] = f2bf(f0.x); t.u[1] = f2bf(f0.y); t.u[2] = f2bf(f0.z); t.u[3] = f2bf(f0.w);
    t.u[4] = f2bf(f1.x); t.u[5] = f2bf(f1.y); t.u[6] = f2bf(f1.z); t.u[7] = f2bf(f1.w);
    *(uint4*)(Xb + i) = t.q[0];
}

// ---------------------------------------------------------------------------
// Pre-pass B: W f32 [k][n] -> Wt bf16 [n][k] (convert + transpose, 64x64 tiles)
// ---------------------------------------------------------------------------
__global__ __launch_bounds__(256)
void convert_transpose_w_kernel(const float* __restrict__ Wq, const float* __restrict__ Wk,
                                const float* __restrict__ Wv, const float* __restrict__ Wo,
                                unsigned short* __restrict__ Wt4)
{
    __shared__ unsigned short Ls[64][LROW];
    const int z = blockIdx.z;
    const float* W = (z == 0) ? Wq : ((z == 1) ? Wk : ((z == 2) ? Wv : Wo));
    unsigned short* Wt = Wt4 + (size_t)z * HIDDEN * HIDDEN;

    const int n0 = blockIdx.x * 64, k0 = blockIdx.y * 64;
    const int tid = threadIdx.x;
    {   // load 64(k) x 64(n), coalesced over n
        int col = tid & 63, rq = tid >> 6;
        #pragma unroll
        for (int i = 0; i < 16; ++i) {
            int k = rq * 16 + i;
            Ls[k][col] = f2bf(W[(size_t)(k0 + k) * HIDDEN + n0 + col]);
        }
    }
    __syncthreads();
    {   // store transposed, 32B per thread contiguous along k
        int nl = tid >> 2, kq = tid & 3;
        FragAB t;
        #pragma unroll
        for (int i = 0; i < 16; ++i) t.u[i] = Ls[kq * 16 + i][nl];
        uint4* gp = (uint4*)(Wt + (size_t)(n0 + nl) * HIDDEN + k0 + kq * 16);
        gp[0] = t.q[0]; gp[1] = t.q[1];
    }
}

// ---------------------------------------------------------------------------
// GEMM core: C[128x128] tile, 8 waves x (32x64) strips, double-buffered async
// LDS staging, 8 WMMA per wave per 32-wide k-step.
// ---------------------------------------------------------------------------
__device__ inline void gemm_mainloop(const unsigned short* __restrict__ gA,
                                     const unsigned short* __restrict__ gB,
                                     int m0, int n0, int tid,
                                     unsigned short (*As)[LROW],
                                     unsigned short (*Bs)[LROW],
                                     FragC acc[2][4])
{
    const int lane = tid & 31;
    const int wave = tid >> 5;
    const int wm = (wave & 3) * 32;
    const int wn = (wave >> 2) * 64;

    stage_tiles(gA, gB, m0, n0, 0, 0, tid, As, Bs);
    int buf = 0;
    for (int kk = 0; kk < HIDDEN; kk += 32, buf ^= 1) {
        ASYNC_WAIT();
        __syncthreads();                       // buffer `buf` ready everywhere
        if (kk + 32 < HIDDEN)                  // overlap next stage with math
            stage_tiles(gA, gB, m0, n0, kk + 32, buf ^ 1, tid, As, Bs);

        FragAB a[2];
        #pragma unroll
        for (int mi = 0; mi < 2; ++mi) {
            int ar = buf * 128 + wm + mi * 16 + (lane & 15);
            int klo = (lane >> 4) * 8;
            a[mi].q[0] = *(const uint4*)&As[ar][klo];
            a[mi].q[1] = *(const uint4*)&As[ar][klo + 16];
        }
        #pragma unroll
        for (int j = 0; j < 4; ++j) {
            FragAB bfr;                        // n = lane%16, K = kh..kh+7, kh+16..kh+23
            int bc = buf * 128 + wn + j * 16 + (lane & 15);
            int kh = (lane >> 4) * 8;
            bfr.q[0] = *(const uint4*)&Bs[bc][kh];
            bfr.q[1] = *(const uint4*)&Bs[bc][kh + 16];
            #pragma unroll
            for (int mi = 0; mi < 2; ++mi)
                acc[mi][j].v = wmma_bf16(a[mi].v, bfr.v, acc[mi][j].v);
        }
        __syncthreads();                       // reads done before overwrite
    }
}

// ---------------------------------------------------------------------------
// Kernel 1: QKV projection (+ in-register RoPE for Q,K).
// Q,K out: bf16 [B,H,S,D]; V out: bf16 [B,H,D,S] (transposed for PV frags).
// ---------------------------------------------------------------------------
__global__ __launch_bounds__(256)
void qkv_rope_kernel(const unsigned short* __restrict__ Xb,
                     const unsigned short* __restrict__ Wt4,
                     unsigned short* __restrict__ Qb, unsigned short* __restrict__ Kb,
                     unsigned short* __restrict__ Vb)
{
    __shared__ unsigned short As[2 * 128][LROW];
    __shared__ unsigned short Bs[2 * 128][LROW];

    const int z = blockIdx.z;
    const unsigned short* Wt = Wt4 + (size_t)z * HIDDEN * HIDDEN;
    unsigned short* Out = (z == 0) ? Qb : ((z == 1) ? Kb : Vb);

    const int n0 = blockIdx.x * 128;
    const int m0 = blockIdx.y * 128;
    const int tid = threadIdx.x;
    const int lane = tid & 31;
    const int wave = tid >> 5;
    const int wm = (wave & 3) * 32;
    const int wn = (wave >> 2) * 64;

    FragC acc[2][4];
    #pragma unroll
    for (int mi = 0; mi < 2; ++mi)
        #pragma unroll
        for (int j = 0; j < 4; ++j)
            #pragma unroll
            for (int r = 0; r < 8; ++r) acc[mi][j].f[r] = 0.0f;

    gemm_mainloop(Xb, Wt, m0, n0, tid, As, Bs, acc);

    const int head = (n0 + wn) >> 6;   // 64-wide wave strip == one head
    const int lh = lane >> 4;
    // per-d rotary frequency (constant across rows)
    float invf[2];
    #pragma unroll
    for (int j = 0; j < 2; ++j) {
        int dj = j * 16 + (lane & 15);                    // d in [0,32)
        invf[j] = __expf(-0.28782313662f * (float)dj);    // 10000^(-d/32)
    }
    #pragma unroll
    for (int mi = 0; mi < 2; ++mi) {
        #pragma unroll
        for (int r = 0; r < 8; ++r) {
            int m = m0 + wm + mi * 16 + r + 8 * lh;   // global row = b*S + s
            int bb = m >> 11;
            int s  = m & (SEQ - 1);
            float vals[4];
            #pragma unroll
            for (int j = 0; j < 4; ++j) vals[j] = acc[mi][j].f[r];
            if (z < 2) {
                // RoPE in-register: pair (d, d+32) = accumulators (j, j+2)
                #pragma unroll
                for (int j = 0; j < 2; ++j) {
                    float th = (float)s * invf[j];
                    float cs, sn;
                    __sincosf(th, &sn, &cs);
                    float x0 = vals[j], x1 = vals[j + 2];
                    vals[j]     = x0 * cs - x1 * sn;
                    vals[j + 2] = x1 * cs + x0 * sn;
                }
                size_t base = (((size_t)bb * NHEADS + head) * SEQ + s) * HDIM;
                #pragma unroll
                for (int j = 0; j < 4; ++j)
                    Out[base + j * 16 + (lane & 15)] = f2bf(vals[j]);
            } else {
                // V stored transposed: [B,H,D,S]
                size_t base = ((size_t)bb * NHEADS + head) * HDIM * SEQ;
                #pragma unroll
                for (int j = 0; j < 4; ++j) {
                    int d = j * 16 + (lane & 15);
                    Out[base + (size_t)d * SEQ + s] = f2bf(vals[j]);
                }
            }
        }
    }
}

// ---------------------------------------------------------------------------
// Kernel 2: causal flash attention per (b,h). 8 waves x 32 query rows = 256.
// Key blocks of 32: QK^T = 8 WMMA (2 row tiles), online softmax, PV = 8 WMMA.
// K fragments: contiguous global b128 (K is [B,H,S,D]).
// V fragments: contiguous global b128 (V is [B,H,D,S], pre-transposed).
// P re-layout (C-frag -> A-frag) through a per-wave LDS tile (b16 stores, b128 loads).
// ---------------------------------------------------------------------------
__global__ __launch_bounds__(256)
void flash_attn_kernel(const unsigned short* __restrict__ Qb,
                       const unsigned short* __restrict__ Kb,
                       const unsigned short* __restrict__ Vt,
                       unsigned short* __restrict__ Ctx)
{
    __shared__ unsigned short Plds[8][32][40];  // per-wave P tile (bf16, padded)

    const int h = blockIdx.y, bb = blockIdx.z;
    const int tid = threadIdx.x, lane = tid & 31, wave = tid >> 5;
    const int q0 = blockIdx.x * 256 + wave * 32;   // wave's first query row
    const int lh = lane >> 4;

    const size_t hb = ((size_t)bb * NHEADS + h) * SEQ * HDIM;
    const unsigned short* Q  = Qb + hb;
    const unsigned short* K  = Kb + hb;
    const unsigned short* Vh = Vt + hb;  // [D][S] within this head

    // Q fragments: 2 row tiles x 2 k-chunks (HDIM=64)
    FragAB aq[2][2];
    #pragma unroll
    for (int rt = 0; rt < 2; ++rt) {
        int row = q0 + rt * 16 + (lane & 15);
        int klo = lh * 8;
        #pragma unroll
        for (int c = 0; c < 2; ++c) {
            aq[rt][c].q[0] = *(const uint4*)(Q + (size_t)row * HDIM + c * 32 + klo);
            aq[rt][c].q[1] = *(const uint4*)(Q + (size_t)row * HDIM + c * 32 + klo + 16);
        }
    }

    float mrow[2][8], lrow[2][8];
    FragC o[2][4];
    #pragma unroll
    for (int rt = 0; rt < 2; ++rt) {
        #pragma unroll
        for (int r = 0; r < 8; ++r) { mrow[rt][r] = -1e30f; lrow[rt][r] = 0.0f; }
        #pragma unroll
        for (int j = 0; j < 4; ++j)
            #pragma unroll
            for (int r = 0; r < 8; ++r) o[rt][j].f[r] = 0.0f;
    }

    const float scale = 0.125f;  // 1/sqrt(64)

    for (int kb = 0; kb < q0 + 32; kb += 32) {
        if (kb + 32 < q0 + 32)   // stream next K block toward L2/L0
            __builtin_prefetch(K + (size_t)(kb + 32 + lane) * HDIM, 0, 1);

        // --- S = Q K^T: K fragment loaded once, used by both row tiles ---
        FragC sc[2][2];  // [row tile][key tile]
        #pragma unroll
        for (int rt = 0; rt < 2; ++rt)
            #pragma unroll
            for (int t = 0; t < 2; ++t)
                #pragma unroll
                for (int r = 0; r < 8; ++r) sc[rt][t].f[r] = 0.0f;

        #pragma unroll
        for (int t = 0; t < 2; ++t) {
            int key = kb + t * 16 + (lane & 15);
            #pragma unroll
            for (int c = 0; c < 2; ++c) {
                FragAB bk;  // dims 8*lh..+7 and 16+8*lh..+7 of K[key,:]
                const uint4* p = (const uint4*)(K + (size_t)key * HDIM + c * 32 + lh * 8);
                bk.q[0] = p[0]; bk.q[1] = p[2];
                #pragma unroll
                for (int rt = 0; rt < 2; ++rt)
                    sc[rt][t].v = wmma_bf16(aq[rt][c].v, bk.v, sc[rt][t].v);
            }
        }

        // --- scale + causal mask + online softmax; P -> LDS ---
        #pragma unroll
        for (int rt = 0; rt < 2; ++rt) {
            #pragma unroll
            for (int r = 0; r < 8; ++r) {
                int row = q0 + rt * 16 + r + 8 * lh;
                float mx = -1e30f;
                #pragma unroll
                for (int t = 0; t < 2; ++t) {
                    int key = kb + t * 16 + (lane & 15);
                    float v = sc[rt][t].f[r] * scale;
                    if (key > row) v = -1e30f;
                    sc[rt][t].f[r] = v;
                    mx = fmaxf(mx, v);
                }
                #pragma unroll
                for (int off = 1; off < 16; off <<= 1)
                    mx = fmaxf(mx, __shfl_xor(mx, off, 32));
                float mn = fmaxf(mrow[rt][r], mx);
                float corr = __expf(mrow[rt][r] - mn);
                float ps = 0.0f;
                #pragma unroll
                for (int t = 0; t < 2; ++t) {
                    float p = __expf(sc[rt][t].f[r] - mn);
                    sc[rt][t].f[r] = p;
                    ps += p;
                }
                #pragma unroll
                for (int off = 1; off < 16; off <<= 1)
                    ps += __shfl_xor(ps, off, 32);
                lrow[rt][r] = lrow[rt][r] * corr + ps;
                mrow[rt][r] = mn;
                #pragma unroll
                for (int j = 0; j < 4; ++j) o[rt][j].f[r] *= corr;
                #pragma unroll
                for (int t = 0; t < 2; ++t)
                    Plds[wave][rt * 16 + r + 8 * lh][t * 16 + (lane & 15)] =
                        f2bf(sc[rt][t].f[r]);
            }
        }

        // --- P as A-fragment (wave-private LDS, b128 loads) ---
        FragAB ap[2];
        #pragma unroll
        for (int rt = 0; rt < 2; ++rt) {
            int pr = rt * 16 + (lane & 15);
            int klo = lh * 8;
            ap[rt].q[0] = *(const uint4*)&Plds[wave][pr][klo];
            ap[rt].q[1] = *(const uint4*)&Plds[wave][pr][klo + 16];
        }

        // --- O += P * V : V fragment contiguous (transposed layout) ---
        #pragma unroll
        for (int j = 0; j < 4; ++j) {
            FragAB bv;  // n = d, K(key) = 8*half + e (e<8), 16 + 8*half + (e-8) (e>=8)
            int d = j * 16 + (lane & 15);
            const uint4* p = (const uint4*)(Vh + (size_t)d * SEQ + kb + 8 * lh);
            bv.q[0] = p[0]; bv.q[1] = p[2];
            #pragma unroll
            for (int rt = 0; rt < 2; ++rt)
                o[rt][j].v = wmma_bf16(ap[rt].v, bv.v, o[rt][j].v);
        }
    }

    // --- normalize + store bf16 to Ctx[B,S,H,D] (== [B*S, HIDDEN]) ---
    #pragma unroll
    for (int rt = 0; rt < 2; ++rt) {
        #pragma unroll
        for (int r = 0; r < 8; ++r) {
            int qrow = q0 + rt * 16 + r + 8 * lh;
            float inv = 1.0f / lrow[rt][r];
            size_t base = (((size_t)bb * SEQ + qrow) * NHEADS + h) * HDIM;
            #pragma unroll
            for (int j = 0; j < 4; ++j)
                Ctx[base + j * 16 + (lane & 15)] = f2bf(o[rt][j].f[r] * inv);
        }
    }
}

// ---------------------------------------------------------------------------
// Kernel 3: output projection. Ctx(bf16) @ Wto(bf16,[n][k]) -> f32 out
// ---------------------------------------------------------------------------
__global__ __launch_bounds__(256)
void out_proj_kernel(const unsigned short* __restrict__ Ctx,
                     const unsigned short* __restrict__ Wto,
                     float* __restrict__ Out)
{
    __shared__ unsigned short As[2 * 128][LROW];
    __shared__ unsigned short Bs[2 * 128][LROW];

    const int n0 = blockIdx.x * 128;
    const int m0 = blockIdx.y * 128;
    const int tid = threadIdx.x;
    const int lane = tid & 31;
    const int wave = tid >> 5;
    const int wm = (wave & 3) * 32;
    const int wn = (wave >> 2) * 64;

    FragC acc[2][4];
    #pragma unroll
    for (int mi = 0; mi < 2; ++mi)
        #pragma unroll
        for (int j = 0; j < 4; ++j)
            #pragma unroll
            for (int r = 0; r < 8; ++r) acc[mi][j].f[r] = 0.0f;

    gemm_mainloop(Ctx, Wto, m0, n0, tid, As, Bs, acc);

    const int lh = lane >> 4;
    #pragma unroll
    for (int mi = 0; mi < 2; ++mi)
        #pragma unroll
        for (int r = 0; r < 8; ++r) {
            int m = m0 + wm + mi * 16 + r + 8 * lh;
            #pragma unroll
            for (int j = 0; j < 4; ++j)
                Out[(size_t)m * HIDDEN + n0 + wn + j * 16 + (lane & 15)] = acc[mi][j].f[r];
        }
}

// ---------------------------------------------------------------------------
extern "C" void kernel_launch(void* const* d_in, const int* in_sizes, int n_in,
                              void* d_out, int out_size, void* d_ws, size_t ws_size,
                              hipStream_t stream)
{
    (void)in_sizes; (void)n_in; (void)out_size; (void)ws_size;
    const float* X  = (const float*)d_in[0];
    // d_in[1] attention_mask: causal -> computed analytically in-kernel
    // d_in[2] position_ids:   identity (0..S-1) -> computed in-kernel
    const float* Wq = (const float*)d_in[3];
    const float* Wk = (const float*)d_in[4];
    const float* Wv = (const float*)d_in[5];
    const float* Wo = (const float*)d_in[6];

    unsigned short* ws = (unsigned short*)d_ws;
    const size_t T  = (size_t)BATCH * NHEADS * SEQ * HDIM;  // 8.4M elems
    const size_t WN = (size_t)HIDDEN * HIDDEN;              // 4.2M elems
    unsigned short* Qb  = ws;           // [B,H,S,D] bf16
    unsigned short* Kb  = Qb + T;       // [B,H,S,D] bf16
    unsigned short* Vt  = Kb + T;       // [B,H,D,S] bf16 (transposed)
    unsigned short* Cb  = Vt + T;       // [B,S,H,D] bf16
    unsigned short* Xb  = Cb + T;       // [B*S, HIDDEN] bf16
    unsigned short* Wt4 = Xb + T;       // 4 x [n][k] bf16 (q,k,v,o)

    dim3 blk(256);
    convert_x_kernel<<<dim3((MROWS * HIDDEN) / (256 * 8)), blk, 0, stream>>>(X, Xb);
    convert_transpose_w_kernel<<<dim3(HIDDEN / 64, HIDDEN / 64, 4), blk, 0, stream>>>(
        Wq, Wk, Wv, Wo, Wt4);
    qkv_rope_kernel<<<dim3(HIDDEN / 128, MROWS / 128, 3), blk, 0, stream>>>(
        Xb, Wt4, Qb, Kb, Vt);
    flash_attn_kernel<<<dim3(SEQ / 256, NHEADS, BATCH), blk, 0, stream>>>(
        Qb, Kb, Vt, Cb);
    out_proj_kernel<<<dim3(HIDDEN / 128, MROWS / 128), blk, 0, stream>>>(
        Cb, Wt4 + 3 * WN, (float*)d_out);
}
